// ProAlignLoss_60842506715329
// MI455X (gfx1250) — matrix operand order repaired and checked by reading, not compiled, not run on hardware
//
#include <hip/hip_runtime.h>
#include <hip/hip_bf16.h>

typedef __attribute__((ext_vector_type(16))) _Float16 v16h;
typedef __attribute__((ext_vector_type(8)))  _Float16 v8h;
typedef __attribute__((ext_vector_type(8)))  float    v8f;

#define B_    8
#define C_    128
#define HW_   (512 * 512)
#define K_    64
#define BK_   512          // B_*K_ total queries
#define NNEG_ 7
#define P_    4096
#define TAU_  0.07f

// ---------------------------------------------------------------------------
// Kernel 0: zero the scalar output (atomicAdd target)
// ---------------------------------------------------------------------------
__global__ void pal_zero(float* out) {
    if (threadIdx.x == 0 && blockIdx.x == 0) out[0] = 0.0f;
}

// ---------------------------------------------------------------------------
// Kernel 1: gather sampled query pixels, L2-normalize, emit f32 + f16 copies.
// One wave32 per query (4 channels/lane), 4 queries per 128-thread block.
// ---------------------------------------------------------------------------
__global__ void pal_gather_norm_q(const float* __restrict__ qfeat,
                                  const int*   __restrict__ qidx,
                                  float*       __restrict__ qn,
                                  _Float16*    __restrict__ qh) {
    const int wave = threadIdx.x >> 5;
    const int lane = threadIdx.x & 31;
    const int qi   = blockIdx.x * 4 + wave;      // 0..511
    const int b    = qi >> 6;                    // /K_
    const int k    = qi & 63;
    const int idx  = qidx[b * K_ + k];
    const float* src = qfeat + ((long long)b * C_) * (long long)HW_ + idx;

    float v[4];
    float ss = 0.0f;
#pragma unroll
    for (int j = 0; j < 4; ++j) {
        const int c = lane + 32 * j;
        v[j] = src[(long long)c * HW_];          // strided gather (1 MB apart)
        ss += v[j] * v[j];
    }
#pragma unroll
    for (int m = 16; m >= 1; m >>= 1) ss += __shfl_xor(ss, m, 32);
    const float inv = 1.0f / fmaxf(sqrtf(ss), 1e-12f);
#pragma unroll
    for (int j = 0; j < 4; ++j) {
        const int c = lane + 32 * j;
        const float t = v[j] * inv;
        qn[qi * C_ + c] = t;
        qh[qi * C_ + c] = (_Float16)t;
    }
}

// ---------------------------------------------------------------------------
// Kernel 2: L2-normalize prototype pool rows -> f16. One wave per row.
// ---------------------------------------------------------------------------
__global__ void pal_norm_pool(const float* __restrict__ pool,
                              _Float16*    __restrict__ ph) {
    const int wave = threadIdx.x >> 5;
    const int lane = threadIdx.x & 31;
    const int row  = blockIdx.x * 4 + wave;      // 0..4095
    const float* src = pool + (long long)row * C_;

    float v[4];
    float ss = 0.0f;
#pragma unroll
    for (int j = 0; j < 4; ++j) {
        v[j] = src[lane + 32 * j];
        ss += v[j] * v[j];
    }
#pragma unroll
    for (int m = 16; m >= 1; m >>= 1) ss += __shfl_xor(ss, m, 32);
    const float inv = 1.0f / fmaxf(sqrtf(ss), 1e-12f);
#pragma unroll
    for (int j = 0; j < 4; ++j)
        ph[row * C_ + lane + 32 * j] = (_Float16)(v[j] * inv);
}

// ---------------------------------------------------------------------------
// Kernel 3: sim = q_f16 @ pool_f16^T via v_wmma_f32_16x16x32_f16, fused
// row-max (== positive logit * TAU). Block = 8 waves, one 16-query M-tile
// per block. N-loop unrolled x2 with two independent accumulator chains so
// the scheduler can interleave WMMAs / loads and avoid the WMMA->VALU
// hazard NOPs. Each wave covers tile pairs {2w,2w+1}, {2w+16,2w+17}, ...
//
// A frag (16x32 f16): lane l -> M = l&15; hi = l>>4 selects K 0..15
//   vs 16..31 (one contiguous 16-half chunk).
// B frag (32x16 f16): lane l -> N = l&15; hi selects K 0..15 vs 16..31
//   (one contiguous 16-half chunk).
// D (16x16 f32): VGPR j, lane l -> M = j + 8*hi, N = l&15.
// ---------------------------------------------------------------------------
__global__ void pal_sim_rowmax(const _Float16* __restrict__ qh,
                               const _Float16* __restrict__ ph,
                               float*          __restrict__ maxsim) {
    __shared__ float smax[16][8];
    const int wave  = threadIdx.x >> 5;
    const int lane  = threadIdx.x & 31;
    const int m     = lane & 15;
    const int hi    = lane >> 4;
    const int mtile = blockIdx.x;                // 0..31

    // Preload A fragments for all 4 K-steps (K = 128 = 4 x 32).
    v16h afrag[4];
    const _Float16* arow = qh + (mtile * 16 + m) * C_;
#pragma unroll
    for (int kk = 0; kk < 4; ++kk) {
        v16h a = *(const v16h*)(arow + kk * 32 + hi * 16);
        afrag[kk] = a;
    }

    float rowmax[8];
#pragma unroll
    for (int j = 0; j < 8; ++j) rowmax[j] = -3.0e38f;

    // 256 N-tiles total; each wave does 32 of them, two per iteration.
    for (int nt = 2 * wave; nt < P_ / 16; nt += 16) {
        const _Float16* brow0 = ph + ((nt + 0) * 16 + m) * C_;  // column N = m
        const _Float16* brow1 = ph + ((nt + 1) * 16 + m) * C_;
        __builtin_prefetch(ph + ((nt + 16) * 16 + m) * C_, 0, 1);
        __builtin_prefetch(ph + ((nt + 17) * 16 + m) * C_, 0, 1);
        v8f acc0 = {};
        v8f acc1 = {};
#pragma unroll
        for (int kk = 0; kk < 4; ++kk) {
            v16h b0 = *(const v16h*)(brow0 + kk * 32 + hi * 16);
            v16h b1 = *(const v16h*)(brow1 + kk * 32 + hi * 16);
            acc0 = __builtin_amdgcn_wmma_f32_16x16x32_f16(
                       false, afrag[kk], false, b0, (short)0, acc0, false, false);
            acc1 = __builtin_amdgcn_wmma_f32_16x16x32_f16(
                       false, afrag[kk], false, b1, (short)0, acc1, false, false);
        }
#pragma unroll
        for (int j = 0; j < 8; ++j)
            rowmax[j] = fmaxf(rowmax[j], fmaxf(acc0[j], acc1[j]));
    }

    // Max over the 16 N-columns held by each half-wave group.
#pragma unroll
    for (int off = 1; off < 16; off <<= 1) {
#pragma unroll
        for (int j = 0; j < 8; ++j)
            rowmax[j] = fmaxf(rowmax[j], __shfl_xor(rowmax[j], off, 32));
    }
    if (m == 0) {
#pragma unroll
        for (int j = 0; j < 8; ++j) smax[j + 8 * hi][wave] = rowmax[j];
    }
    __syncthreads();
    if (threadIdx.x < 16) {
        float v = smax[threadIdx.x][0];
#pragma unroll
        for (int w = 1; w < 8; ++w) v = fmaxf(v, smax[threadIdx.x][w]);
        maxsim[mtile * 16 + threadIdx.x] = v;    // = q . pos_proto (cosine)
    }
}

// ---------------------------------------------------------------------------
// Kernel 4: per-query InfoNCE loss. Wave per query; negatives normalized on
// the fly (fused dot & sum-sq reductions). Stable log-sum-exp; mean via
// atomicAdd of loss/512.
// ---------------------------------------------------------------------------
__global__ void pal_loss(const float* __restrict__ qn,
                         const float* __restrict__ negs,
                         const float* __restrict__ maxsim,
                         float*       __restrict__ out) {
    const int wave = threadIdx.x >> 5;
    const int lane = threadIdx.x & 31;
    const int qi   = blockIdx.x * 4 + wave;

    const float* q = qn + qi * C_;
    float qv[4];
#pragma unroll
    for (int j = 0; j < 4; ++j) qv[j] = q[lane + 32 * j];

    float logit[8];
    logit[0] = maxsim[qi] / TAU_;
    for (int n = 0; n < NNEG_; ++n) {
        const float* nv = negs + ((long long)qi * NNEG_ + n) * C_;
        float dot = 0.0f, sq = 0.0f;
#pragma unroll
        for (int j = 0; j < 4; ++j) {
            const float x = nv[lane + 32 * j];
            dot += qv[j] * x;
            sq  += x * x;
        }
#pragma unroll
        for (int mm = 16; mm >= 1; mm >>= 1) {
            dot += __shfl_xor(dot, mm, 32);
            sq  += __shfl_xor(sq,  mm, 32);
        }
        logit[n + 1] = dot / (fmaxf(sqrtf(sq), 1e-12f) * TAU_);
    }

    if (lane == 0) {
        float mx = logit[0];
#pragma unroll
        for (int n = 1; n < 8; ++n) mx = fmaxf(mx, logit[n]);
        float s = 0.0f;
#pragma unroll
        for (int n = 0; n < 8; ++n) s += __expf(logit[n] - mx);
        const float loss = logf(s) + mx - logit[0];
        atomicAdd(out, loss * (1.0f / (float)BK_));
    }
}

// ---------------------------------------------------------------------------
extern "C" void kernel_launch(void* const* d_in, const int* in_sizes, int n_in,
                              void* d_out, int out_size, void* d_ws, size_t ws_size,
                              hipStream_t stream) {
    const float* qfeat = (const float*)d_in[0];   // [B, C, H, W]
    const float* pool  = (const float*)d_in[1];   // [P, C]
    const float* negs  = (const float*)d_in[2];   // [B, K, NNEG, C]
    const int*   qidx  = (const int*)  d_in[3];   // [B, K]
    float* out = (float*)d_out;

    // Workspace layout (bytes):
    //   [0,       262144) : q_norm f32   512 x 128
    //   [262144,  393216) : q       f16  512 x 128
    //   [393216, 1441792) : pool    f16 4096 x 128
    //   [1441792,1443840) : maxsim  f32  512
    char* ws = (char*)d_ws;
    float*    qn     = (float*)   (ws + 0);
    _Float16* qh     = (_Float16*)(ws + 262144);
    _Float16* ph     = (_Float16*)(ws + 393216);
    float*    maxsim = (float*)   (ws + 1441792);

    pal_zero<<<1, 32, 0, stream>>>(out);
    pal_gather_norm_q<<<BK_ / 4, 128, 0, stream>>>(qfeat, qidx, qn, qh);
    pal_norm_pool<<<P_ / 4, 128, 0, stream>>>(pool, ph);
    pal_sim_rowmax<<<BK_ / 16, 256, 0, stream>>>(qh, ph, maxsim);
    pal_loss<<<BK_ / 4, 128, 0, stream>>>(qn, negs, maxsim, out);
}
